// FrameQuantizer_11879879544491
// MI455X (gfx1250) — compile-verified, matrix operand order checked by reading, not run on hardware
//
#include <hip/hip_runtime.h>
#include <hip/hip_bf16.h>
#include <stdint.h>

// Problem dimensions (fixed by the reference)
#define BB 4
#define CC 8
#define HH 128
#define WW 2048
#define TT (BB * WW)   // 8192 tokens
#define DD (CC * HH)   // 1024 token dim
#define NEMB 8192      // codebook entries
#define BETA 0.25f

// GEMM tiling
#define BMt 64          // block tile M
#define BNt 256         // block tile N
#define KC 32           // K slab per step
#define KPAD 8          // pad LDS rows to 40 ushorts (80 B) -> conflict-free b128 reads
#define LROW (KC + KPAD)
#define NCHUNKS 4
#define NCHUNK (NEMB / NCHUNKS)   // 2048

typedef __attribute__((ext_vector_type(16))) __bf16 v16bf;
typedef __attribute__((ext_vector_type(8)))  float  v8f;
typedef __attribute__((ext_vector_type(8)))  unsigned short us8;

__device__ __forceinline__ unsigned short f2bf(float f) {
  unsigned u = __float_as_uint(f);
  u += 0x7FFFu + ((u >> 16) & 1u);   // round-to-nearest-even bf16
  return (unsigned short)(u >> 16);
}

// Monotonic float->uint mapping so u64 min == (min distance, then min index)
__device__ __forceinline__ unsigned long long pack_key(float v, unsigned idx) {
  unsigned u = __float_as_uint(v);
  u = (u & 0x80000000u) ? ~u : (u | 0x80000000u);
  return ((unsigned long long)u << 32) | (unsigned long long)idx;
}

// Flat shared pointer -> wave-relative LDS byte offset (addr[31:0] per aperture map)
__device__ __forceinline__ unsigned lds_off_of(const void* p) {
  return (unsigned)(size_t)p;
}

// CDNA5 async global -> LDS copy, 16 B per lane, tracked by ASYNCcnt
__device__ __forceinline__ void async_b128(unsigned lds_off, unsigned long long gaddr) {
  asm volatile("global_load_async_to_lds_b128 %0, %1, off"
               :: "v"(lds_off), "v"(gaddr) : "memory");
}
__device__ __forceinline__ void wait_async0() {
  asm volatile("s_wait_asynccnt 0x0" ::: "memory");
}

// ---------------- kernel 1: init keys + loss slot ----------------
__global__ void vq_init(unsigned long long* __restrict__ keys,
                        float* __restrict__ loss_slot) {
  int i = blockIdx.x * blockDim.x + threadIdx.x;
  if (i < TT) keys[i] = ~0ull;
  if (i == 0) *loss_slot = 0.0f;
}

// ---------------- kernel 2: (b,c,h,w) -> token-major bf16 ----------------
__global__ void vq_prep_z(const float* __restrict__ z,
                          unsigned short* __restrict__ zbf) {
  int i = blockIdx.x * 256 + threadIdx.x;   // i == source flat index, coalesced read
  int w  = i & (WW - 1);
  int ch = i >> 11;                          // WW = 2048
  int h  = ch & (HH - 1);
  int bc = ch >> 7;                          // HH = 128
  int c  = bc & (CC - 1);
  int b  = bc >> 3;                          // CC = 8
  int t = b * WW + w;
  int d = c * HH + h;
  zbf[(size_t)t * DD + d] = f2bf(z[i]);
}

// ---------------- kernel 3: codebook -> bf16 + ||e||^2 ----------------
__global__ void vq_prep_e(const float* __restrict__ e,
                          unsigned short* __restrict__ ebf,
                          float* __restrict__ enorm) {
  __shared__ float red[256];
  const int n = blockIdx.x;
  const int tid = threadIdx.x;
  const float* row = e + (size_t)n * DD;
  float s = 0.0f;
  for (int d = tid; d < DD; d += 256) {
    float v = row[d];
    ebf[(size_t)n * DD + d] = f2bf(v);
    s += v * v;
  }
  red[tid] = s; __syncthreads();
  for (int k = 128; k > 0; k >>= 1) {
    if (tid < k) red[tid] += red[tid + k];
    __syncthreads();
  }
  if (tid == 0) enorm[n] = red[0];
}

// ---------------- kernel 4: WMMA distance + argmin ----------------
__global__ __launch_bounds__(256) void vq_dist_argmin(
    const unsigned short* __restrict__ zbf,
    const unsigned short* __restrict__ ebf,
    const float* __restrict__ enorm,
    unsigned long long* __restrict__ keys) {
  // Double-buffered, padded slabs: A 2*64*40, B 2*256*40 ushorts (~50 KB)
  __shared__ __align__(16) unsigned short Ab[2][BMt * LROW];
  __shared__ __align__(16) unsigned short Bb[2][BNt * LROW];

  const int tid  = threadIdx.x;
  const int lane = tid & 31;
  const int wave = tid >> 5;            // 8 waves
  const int wm   = wave & 1;            // 2 M wave-tiles of 32
  const int wn   = wave >> 1;           // 4 N wave-tiles of 64
  const int fr   = lane & 15;           // row/col within 16x16 tile
  const int khalf = lane >> 4;          // 0: K 0-7/16-23, 1: K 8-15/24-31

  const int m_base  = blockIdx.x * BMt;
  const int nc_base = blockIdx.y * NCHUNK;

  // staging assignment (per thread per K slab): A one b128, B four b128
  const int st_row = tid >> 2, st_ch = tid & 3;
  const unsigned a_lds[2] = {
    lds_off_of(&Ab[0][st_row * LROW + st_ch * 8]),
    lds_off_of(&Ab[1][st_row * LROW + st_ch * 8])};
  unsigned b_lds[4][2];
#pragma unroll
  for (int s = 0; s < 4; ++s) {
    b_lds[s][0] = lds_off_of(&Bb[0][(st_row + s * 64) * LROW + st_ch * 8]);
    b_lds[s][1] = lds_off_of(&Bb[1][(st_row + s * 64) * LROW + st_ch * 8]);
  }
  const unsigned long long a_gbase =
      (unsigned long long)(size_t)(zbf + (size_t)(m_base + st_row) * DD + st_ch * 8);

  // fragment read rows (padded stride, 16 B aligned, conflict-free)
  const int ar0 = (wm * 32 + fr) * LROW;
  const int ar1 = (wm * 32 + 16 + fr) * LROW;
  int br[4];
#pragma unroll
  for (int j = 0; j < 4; ++j) br[j] = (wn * 64 + j * 16 + fr) * LROW;

  float    minval[2][8];
  unsigned minidx[2][8];
#pragma unroll
  for (int s = 0; s < 2; ++s)
#pragma unroll
    for (int r = 0; r < 8; ++r) { minval[s][r] = 3.4e38f; minidx[s][r] = 0u; }

  for (int nt = 0; nt < NCHUNK; nt += BNt) {
    const int n_base = nc_base + nt;
    unsigned long long b_gbase[4];
#pragma unroll
    for (int s = 0; s < 4; ++s)
      b_gbase[s] = (unsigned long long)(size_t)
          (ebf + (size_t)(n_base + st_row + s * 64) * DD + st_ch * 8);

    v8f acc[2][4];
#pragma unroll
    for (int i = 0; i < 2; ++i)
#pragma unroll
      for (int j = 0; j < 4; ++j) acc[i][j] = (v8f){0.f,0.f,0.f,0.f,0.f,0.f,0.f,0.f};

    // prologue: slab 0 -> buffer 0
    async_b128(a_lds[0], a_gbase);
#pragma unroll
    for (int s = 0; s < 4; ++s) async_b128(b_lds[s][0], b_gbase[s]);

#pragma unroll 2
    for (int k = 0; k < DD / KC; ++k) {
      const int cur = k & 1;
      wait_async0();        // my slab-k copies landed in LDS
      __syncthreads();      // everyone's landed; everyone done reading buf[cur^1]
      if (k + 1 < DD / KC) {
        const unsigned long long koff = (unsigned long long)((k + 1) * KC) * 2ull;
        async_b128(a_lds[cur ^ 1], a_gbase + koff);
#pragma unroll
        for (int s = 0; s < 4; ++s) async_b128(b_lds[s][cur ^ 1], b_gbase[s] + koff);
      }
      // fragments (ISA 7.12.2): lane 0-15 K{0-7},{16-23}; lane 16-31 K{8-15},{24-31}
      union { us8 h[2]; v16bf v; } af0, af1, bf[4];
      const us8* ap0 = (const us8*)(Ab[cur] + ar0);
      const us8* ap1 = (const us8*)(Ab[cur] + ar1);
      af0.h[0] = ap0[khalf]; af0.h[1] = ap0[2 + khalf];
      af1.h[0] = ap1[khalf]; af1.h[1] = ap1[2 + khalf];
#pragma unroll
      for (int j = 0; j < 4; ++j) {
        const us8* bp = (const us8*)(Bb[cur] + br[j]);
        bf[j].h[0] = bp[khalf]; bf[j].h[1] = bp[2 + khalf];
      }
#pragma unroll
      for (int j = 0; j < 4; ++j) {
        acc[0][j] = __builtin_amdgcn_wmma_f32_16x16x32_bf16(
            false, af0.v, false, bf[j].v, (short)0, acc[0][j], false, false);
        acc[1][j] = __builtin_amdgcn_wmma_f32_16x16x32_bf16(
            false, af1.v, false, bf[j].v, (short)0, acc[1][j], false, false);
      }
    }

    // epilogue: dist = ||e||^2 - 2*dot (row-constant ||z||^2 dropped for argmin)
#pragma unroll
    for (int j = 0; j < 4; ++j) {
      const int n = n_base + wn * 64 + j * 16 + fr;
      const float en = enorm[n];
#pragma unroll
      for (int r = 0; r < 8; ++r) {
        float v;
        v = en - 2.0f * acc[0][j][r];
        if (v < minval[0][r]) { minval[0][r] = v; minidx[0][r] = (unsigned)n; }
        v = en - 2.0f * acc[1][j][r];
        if (v < minval[1][r]) { minval[1][r] = v; minidx[1][r] = (unsigned)n; }
      }
    }
  }

  // reduce over the 16 lanes sharing each output row, then one u64 atomicMin/row
#pragma unroll
  for (int s = 0; s < 2; ++s) {
    unsigned long long key[8];
#pragma unroll
    for (int r = 0; r < 8; ++r) key[r] = pack_key(minval[s][r], minidx[s][r]);
#pragma unroll
    for (int m = 1; m < 16; m <<= 1) {
#pragma unroll
      for (int r = 0; r < 8; ++r) {
        unsigned long long o = __shfl_xor(key[r], m, 32);
        if (o < key[r]) key[r] = o;
      }
    }
    if (fr == 0) {
      // lane 0 holds rows ..+{0..7}; lane 16 holds rows ..+8+{0..7}
      int rowb = m_base + wm * 32 + s * 16 + (khalf << 3);
#pragma unroll
      for (int r = 0; r < 8; ++r)
        atomicMin(&keys[rowb + r], key[r]);
    }
  }
}

// ---------------- kernel 5: gather, scatter to (b,c,h,w), loss, idx ----------------
__global__ void vq_output(const float* __restrict__ z,
                          const float* __restrict__ emb,
                          const unsigned long long* __restrict__ keys,
                          float* __restrict__ out) {
  __shared__ float red[256];
  const int t = blockIdx.x;
  const int tid = threadIdx.x;
  const int idx = (int)(unsigned)(keys[t] & 0xFFFFFFFFull);
  const int b = t >> 11;          // t / WW
  const int w = t & (WW - 1);
  float s = 0.0f;
  for (int d = tid; d < DD; d += 256) {
    int c = d >> 7, h = d & (HH - 1);
    size_t zi = ((size_t)((b * CC + c) * HH + h)) * WW + w;
    float e  = emb[(size_t)idx * DD + d];
    float zv = z[zi];
    out[zi] = e;                  // straight-through forward value == z_q
    float diff = e - zv;
    s += diff * diff;
  }
  red[tid] = s; __syncthreads();
  for (int k = 128; k > 0; k >>= 1) {
    if (tid < k) red[tid] += red[tid + k];
    __syncthreads();
  }
  if (tid == 0) {
    // loss value = (1 + BETA) * mean((z_q - z)^2)
    atomicAdd(out + (size_t)TT * DD,
              red[0] * ((1.0f + BETA) / (float)((size_t)TT * DD)));
    out[(size_t)TT * DD + 1 + t] = (float)idx;
  }
}

extern "C" void kernel_launch(void* const* d_in, const int* in_sizes, int n_in,
                              void* d_out, int out_size, void* d_ws, size_t ws_size,
                              hipStream_t stream) {
  const float* z   = (const float*)d_in[0];   // 4*8*128*2048 f32
  const float* emb = (const float*)d_in[1];   // 8192*1024 f32
  float* out = (float*)d_out;                 // z_q (8388608) | loss (1) | idx (8192)

  char* ws = (char*)d_ws;
  unsigned long long* keys = (unsigned long long*)ws;                      // 64 KB
  float* enorm = (float*)(ws + (size_t)TT * 8);                            // 32 KB
  unsigned short* zbf = (unsigned short*)(ws + (size_t)TT * 8 + (size_t)NEMB * 4);
  unsigned short* ebf = zbf + (size_t)TT * DD;                             // 16 MB each

  vq_init<<<(TT + 255) / 256, 256, 0, stream>>>(keys, out + (size_t)TT * DD);
  vq_prep_z<<<(TT * DD) / 256, 256, 0, stream>>>(z, zbf);
  vq_prep_e<<<NEMB, 256, 0, stream>>>(emb, ebf, enorm);
  dim3 grid(TT / BMt, NCHUNKS);
  vq_dist_argmin<<<grid, 256, 0, stream>>>(zbf, ebf, enorm, keys);
  vq_output<<<TT, 256, 0, stream>>>(z, emb, keys, out);
}
